// SelfAttentionBlock_63788854280229
// MI455X (gfx1250) — matrix operand order repaired and checked by reading, not compiled, not run on hardware
//
#include <hip/hip_runtime.h>
#include <hip/hip_bf16.h>

typedef __attribute__((ext_vector_type(16))) __bf16 v16bf;
typedef __attribute__((ext_vector_type(8)))  float  v8f;
typedef __attribute__((ext_vector_type(4)))  unsigned u32x4;
typedef __attribute__((ext_vector_type(8)))  unsigned u32x8;

union FragU { uint4 q[2]; v16bf v; };

__device__ __forceinline__ v8f wmma_bf16(v16bf a, v16bf b, v8f c) {
    return __builtin_amdgcn_wmma_f32_16x16x32_bf16(
        false, a, false, b, (short)0, c, false, false);
}

// A operand 16x32 bf16: lanes 0-15 row M=lr, K chunks {0..7,16..23};
// lanes 16-31 row M=lr, chunks {8..15,24..31}. row_ptr -> row start, K contiguous.
__device__ __forceinline__ v16bf ld_afrag(const __bf16* row_ptr, int half) {
    FragU f;
    f.q[0] = *(const uint4*)(row_ptr + half * 8);
    f.q[1] = *(const uint4*)(row_ptr + 16 + half * 8);
    return f.v;
}

// B operand 32x16 bf16 from B^T storage (per-column K runs contiguous):
// lanes 0-15 col N=lr hold K chunks {0..7,16..23}; lanes 16-31 col N=lr hold {8..15,24..31}.
__device__ __forceinline__ v16bf ld_bfrag(const __bf16* col_ptr, int half) {
    FragU f;
    f.q[0] = *(const uint4*)(col_ptr + half * 8);
    f.q[1] = *(const uint4*)(col_ptr + 16 + half * 8);
    return f.v;
}

__device__ __forceinline__ unsigned pk2bf(float a, float b) {
    __bf16 ha = (__bf16)a, hb = (__bf16)b;
    unsigned short ua = __builtin_bit_cast(unsigned short, ha);
    unsigned short ub = __builtin_bit_cast(unsigned short, hb);
    return ((unsigned)ub << 16) | (unsigned)ua;
}

// ---------------------------------------------------------------------------
// TDM: async 2D bf16 tile copy global -> LDS with LDS row padding.
//   tile = rows x cols (bf16 elements), global row stride = gstride elements,
//   LDS row stride = cols + 8 elements (pad 16B after each 2B*cols==128B row).
// Issued by one wave; completion via s_wait_tensorcnt + workgroup barrier.
// ---------------------------------------------------------------------------
__device__ __forceinline__ void tdm_load_tile_bf16(
    unsigned lds_off_bytes, const void* gptr,
    unsigned cols, unsigned rows, unsigned gstride_elems)
{
    unsigned long long ga = (unsigned long long)gptr;
    u32x4 g0;
    g0[0] = 1u;                                   // count=1, user mode
    g0[1] = lds_off_bytes;                        // lds_addr
    g0[2] = (unsigned)ga;                         // global_addr[31:0]
    g0[3] = (unsigned)((ga >> 32) & 0x1FFFFFFu)   // global_addr[56:32]
          | (2u << 30);                           // type=2 ("image")
    u32x8 g1;
    // data_size=2B (code 1), pad_enable=1, pad_interval=32 DW (code 4),
    // pad_amount=4 DW (code 3)  -> LDS row stride 144B = 72 bf16
    g1[0] = (1u << 16) | (1u << 20) | (4u << 22) | (3u << 25);
    g1[1] = (cols & 0xFFFFu) << 16;               // tensor_dim0[15:0]
    g1[2] = (rows & 0xFFFFu) << 16;               // dim0 hi=0 | tensor_dim1 lo
    g1[3] = (cols & 0xFFFFu) << 16;               // dim1 hi=0 | tile_dim0
    g1[4] = rows & 0xFFFFu;                       // tile_dim1 | tile_dim2=0
    g1[5] = gstride_elems;                        // tensor_dim0_stride[31:0]
    g1[6] = 0u;                                   // stride hi | dim1_stride lo
    g1[7] = 0u;
    asm volatile("tensor_load_to_lds %0, %1" :: "s"(g0), "s"(g1) : "memory");
}

// ---------------------------------------------------------------------------
// GEMM + bias: C[M,N] = A[M,K] * B[K,N] + bias[N]
// Block tile 128(M) x BN(N), K-step 32. 8 waves, each 16M x BN (BN/16 accs).
// ---------------------------------------------------------------------------
template<int BN, bool A_BF16, bool OUT_F32>
__global__ __launch_bounds__(256)
void gemm_bias_kernel(const void* __restrict__ Ap, const float* __restrict__ Bp,
                      const float* __restrict__ bias, void* __restrict__ Cp,
                      int M, int N, int K)
{
    constexpr int BM = 128, BK = 32, LDA = 40, LDB = 40;
    constexpr int NT = BN / 16;          // WMMA tiles per wave along N
    constexpr int EPB = BN / 8;          // B elements staged per thread
    __shared__ __align__(16) __bf16 As[BM * LDA];   // [m][k] row-major bf16
    __shared__ __align__(16) __bf16 Bs[BN * LDB];   // B^T: [n][k] bf16

    const int t = threadIdx.x;
    const int wave = t >> 5, lane = t & 31, lr = lane & 15, half = lane >> 4;
    const int n0 = blockIdx.x * BN, m0 = blockIdx.y * BM;

    v8f zero8 = (v8f)0.0f;
    v8f acc[NT];
    #pragma unroll
    for (int j = 0; j < NT; ++j) acc[j] = zero8;

    for (int kt = 0; kt < K; kt += BK) {
        // ---- stage A tile (128x32) into LDS as bf16 ----
        {
            const int m = t >> 1, kk = (t & 1) * 16;
            if constexpr (A_BF16) {
                const __bf16* src = (const __bf16*)Ap + (size_t)(m0 + m) * K + kt + kk;
                *(uint4*)&As[m * LDA + kk]     = *(const uint4*)src;
                *(uint4*)&As[m * LDA + kk + 8] = *(const uint4*)(src + 8);
                if (kt + BK < K) __builtin_prefetch(src + BK, 0, 0);
            } else {
                const float* src = (const float*)Ap + (size_t)(m0 + m) * K + kt + kk;
                float4 f0 = *(const float4*)(src + 0);
                float4 f1 = *(const float4*)(src + 4);
                float4 f2 = *(const float4*)(src + 8);
                float4 f3 = *(const float4*)(src + 12);
                uint4 lo = make_uint4(pk2bf(f0.x, f0.y), pk2bf(f0.z, f0.w),
                                      pk2bf(f1.x, f1.y), pk2bf(f1.z, f1.w));
                uint4 hi = make_uint4(pk2bf(f2.x, f2.y), pk2bf(f2.z, f2.w),
                                      pk2bf(f3.x, f3.y), pk2bf(f3.z, f3.w));
                *(uint4*)&As[m * LDA + kk]     = lo;
                *(uint4*)&As[m * LDA + kk + 8] = hi;
                if (kt + BK < K) __builtin_prefetch(src + BK, 0, 0);
            }
        }
        // ---- stage B tile (32xBN) into LDS transposed (B^T[n][k]) ----
        {
            const int kk = t >> 3, nn = (t & 7) * EPB;
            const float* src = Bp + (size_t)(kt + kk) * N + n0 + nn;
            #pragma unroll
            for (int e = 0; e < EPB; e += 4) {
                float4 f = *(const float4*)(src + e);
                Bs[(nn + e + 0) * LDB + kk] = (__bf16)f.x;
                Bs[(nn + e + 1) * LDB + kk] = (__bf16)f.y;
                Bs[(nn + e + 2) * LDB + kk] = (__bf16)f.z;
                Bs[(nn + e + 3) * LDB + kk] = (__bf16)f.w;
            }
        }
        __syncthreads();

        v16bf a = ld_afrag(&As[(wave * 16 + lr) * LDA], half);
        #pragma unroll
        for (int j = 0; j < NT; ++j) {
            v16bf b = ld_bfrag(&Bs[(j * 16 + lr) * LDB], half);
            acc[j] = wmma_bf16(a, b, acc[j]);
        }
        __syncthreads();
    }

    // ---- epilogue: C layout (VGPR r: lanes0-15 -> M=r, lanes16-31 -> M=r+8) ----
    #pragma unroll
    for (int j = 0; j < NT; ++j) {
        const int col = n0 + j * 16 + lr;
        const float bj = bias[col];
        #pragma unroll
        for (int r = 0; r < 8; ++r) {
            const int row = m0 + wave * 16 + r + half * 8;
            const float v = acc[j][r] + bj;
            if constexpr (OUT_F32)
                ((float*)Cp)[(size_t)row * N + col] = v;
            else
                ((__bf16*)Cp)[(size_t)row * N + col] = (__bf16)v;
        }
    }
}

// ---------------------------------------------------------------------------
// Causal flash attention. qkv[B*T][3C] bf16 (k|q|v thirds). One block per
// (128 q-rows, head, batch). 8 waves x 16 q-rows each. K-tile = 32 keys.
// K tile staged via TDM tensor_load_to_lds (async DMA, LDS pad = 72-el rows).
// ---------------------------------------------------------------------------
__global__ __launch_bounds__(256)
void attn_kernel(const __bf16* __restrict__ qkv, __bf16* __restrict__ att)
{
    constexpr int T = 2048, C = 1024, C3 = 3072, HS = 64;
    constexpr int LDK = 72, LDV = 40, LDP = 40;
    __shared__ __align__(16) __bf16 Ks[32 * LDK];      // K tile [key][d]
    __shared__ __align__(16) __bf16 Vt[64 * LDV];      // V^T tile [d][key]
    __shared__ __align__(16) __bf16 Ps[8][16 * LDP];   // per-wave P [q][key]

    const int t = threadIdx.x;
    const int wave = t >> 5, lane = t & 31, lr = lane & 15, half = lane >> 4;
    const int qi0 = blockIdx.x * 128, h = blockIdx.y, b = blockIdx.z;
    const size_t rowbase = (size_t)b * T;
    const unsigned ks_lds = (unsigned)(unsigned long long)(uintptr_t)&Ks[0];

    // Q fragments: loaded directly in A-operand layout from global
    const int qrow_l = qi0 + wave * 16 + lr;
    const __bf16* qptr = qkv + (rowbase + qrow_l) * C3 + C + h * HS;
    FragU fa0, fa1;
    fa0.q[0] = *(const uint4*)(qptr + half * 8);
    fa0.q[1] = *(const uint4*)(qptr + 16 + half * 8);
    fa1.q[0] = *(const uint4*)(qptr + 32 + half * 8);
    fa1.q[1] = *(const uint4*)(qptr + 48 + half * 8);

    v8f zero8 = (v8f)0.0f;
    v8f O[4] = { zero8, zero8, zero8, zero8 };
    float m_run[8], l_run[8];
    #pragma unroll
    for (int r = 0; r < 8; ++r) { m_run[r] = -__builtin_inff(); l_run[r] = 0.0f; }

    const int kmax = qi0 + 128;
    for (int kj = 0; kj < kmax; kj += 32) {
        // ---- K tile [32 keys][64 d] via TDM async DMA (wave 0 issues) ----
        if (wave == 0) {
            const __bf16* ksrc = qkv + (rowbase + kj) * C3 + h * HS;  // k third
            tdm_load_tile_bf16(ks_lds, ksrc, HS, 32, C3);
            __builtin_amdgcn_s_wait_tensorcnt(0);
        }
        // ---- stage V^T tile [64 d][32 keys] (manual: needs transpose) ----
        {
            const int key = t >> 3, d = (t & 7) * 8;
            const __bf16* src = qkv + (rowbase + kj + key) * C3 + 2 * C + h * HS + d;
            uint4 u = *(const uint4*)src;
            const __bf16* e = (const __bf16*)&u;
            #pragma unroll
            for (int j = 0; j < 8; ++j) Vt[(d + j) * LDV + key] = e[j];
        }
        __syncthreads();

        // S = Q K^T  (two 16x16 key tiles, each over d=0..31 and d=32..63)
        v8f s0 = zero8, s1 = zero8;
        s0 = wmma_bf16(fa0.v, ld_bfrag(&Ks[lr * LDK + 0],  half), s0);
        s0 = wmma_bf16(fa1.v, ld_bfrag(&Ks[lr * LDK + 32], half), s0);
        s1 = wmma_bf16(fa0.v, ld_bfrag(&Ks[(16 + lr) * LDK + 0],  half), s1);
        s1 = wmma_bf16(fa1.v, ld_bfrag(&Ks[(16 + lr) * LDK + 32], half), s1);

        // scale + causal mask
        const int key0 = kj + lr, key1 = kj + 16 + lr;
        #pragma unroll
        for (int r = 0; r < 8; ++r) {
            const int qr = qi0 + wave * 16 + r + half * 8;
            float v0 = s0[r] * 0.125f; if (key0 > qr) v0 = -__builtin_inff();
            float v1 = s1[r] * 0.125f; if (key1 > qr) v1 = -__builtin_inff();
            s0[r] = v0; s1[r] = v1;
        }

        // online softmax (per-row stats; rows live in 16-lane halves)
        float scale[8];
        #pragma unroll
        for (int r = 0; r < 8; ++r) {
            float mx = fmaxf(s0[r], s1[r]);
            mx = fmaxf(mx, __shfl_xor(mx, 1, 32));
            mx = fmaxf(mx, __shfl_xor(mx, 2, 32));
            mx = fmaxf(mx, __shfl_xor(mx, 4, 32));
            mx = fmaxf(mx, __shfl_xor(mx, 8, 32));
            const float m_new = fmaxf(m_run[r], mx);
            const float sc = __expf(m_run[r] - m_new);
            const float p0 = __expf(s0[r] - m_new);
            const float p1 = __expf(s1[r] - m_new);
            s0[r] = p0; s1[r] = p1;
            float rs = p0 + p1;
            rs += __shfl_xor(rs, 1, 32);
            rs += __shfl_xor(rs, 2, 32);
            rs += __shfl_xor(rs, 4, 32);
            rs += __shfl_xor(rs, 8, 32);
            l_run[r] = l_run[r] * sc + rs;
            m_run[r] = m_new;
            scale[r] = sc;
        }

        // P: C-layout -> wave-private LDS row-major [16 q][32 key] -> A-operand
        __bf16* pw = Ps[wave];
        #pragma unroll
        for (int r = 0; r < 8; ++r) {
            pw[(r + half * 8) * LDP + lr]      = (__bf16)s0[r];
            pw[(r + half * 8) * LDP + 16 + lr] = (__bf16)s1[r];
        }
        v16bf pa = ld_afrag(&pw[lr * LDP], half);

        v8f scv;
        #pragma unroll
        for (int r = 0; r < 8; ++r) scv[r] = scale[r];

        #pragma unroll
        for (int j = 0; j < 4; ++j) {
            v16bf vb = ld_bfrag(&Vt[(j * 16 + lr) * LDV], half);
            v8f c = O[j] * scv;
            O[j] = wmma_bf16(pa, vb, c);
        }
        __syncthreads();
    }

    // epilogue: out[b, q, h*64 + d] = O / l
    #pragma unroll
    for (int r = 0; r < 8; ++r) {
        const int qr = qi0 + wave * 16 + r + half * 8;
        const float inv = 1.0f / l_run[r];
        #pragma unroll
        for (int j = 0; j < 4; ++j) {
            const int d = j * 16 + lr;
            att[(rowbase + qr) * C + h * HS + d] = (__bf16)(O[j][r] * inv);
        }
    }
}

extern "C" void kernel_launch(void* const* d_in, const int* in_sizes, int n_in,
                              void* d_out, int out_size, void* d_ws, size_t ws_size,
                              hipStream_t stream) {
    const float* x       = (const float*)d_in[0];  // [2,2048,1024]
    const float* W_atten = (const float*)d_in[1];  // [1024,3072]
    const float* b_atten = (const float*)d_in[2];  // [3072]
    const float* W_proj  = (const float*)d_in[3];  // [1024,1024]
    const float* b_proj  = (const float*)d_in[4];  // [1024]

    constexpr int M = 2 * 2048;   // B*T
    constexpr int K = 1024;       // C
    constexpr int N1 = 3072;      // 3C
    constexpr int N2 = 1024;

    __bf16* qkv = (__bf16*)d_ws;                                   // 25.2 MB
    __bf16* attb = (__bf16*)((char*)d_ws + (size_t)M * N1 * 2);    // + 8.4 MB

    // 1) qkv = x @ W_atten + b_atten   (fp32 in, bf16 out) — wide 128x128 tile
    gemm_bias_kernel<128, false, false>
        <<<dim3(N1 / 128, M / 128), 256, 0, stream>>>(x, W_atten, b_atten, qkv, M, N1, K);

    // 2) causal flash attention over qkv (k|q|v thirds), TDM-staged K tiles
    attn_kernel<<<dim3(2048 / 128, 16, 2), 256, 0, stream>>>(qkv, attb);

    // 3) out = att @ W_proj + b_proj   (bf16 in, fp32 out) — 128x64 tile
    gemm_bias_kernel<64, true, true>
        <<<dim3(N2 / 64, M / 128), 256, 0, stream>>>(attb, W_proj, b_proj, (float*)d_out, M, N2, K);
}
